// CompoundProteinInteractionPrediction_16097537425591
// MI455X (gfx1250) — matrix operand order repaired and checked by reading, not compiled, not run on hardware
//
#include <hip/hip_runtime.h>
#include <hip/hip_bf16.h>

// ---------------------------------------------------------------------------
// CDNA5 (gfx1250) implementation: all GEMMs / attention contractions run on
// v_wmma_f32_16x16x32_bf16 (wave32), fp32 accumulate; softmax/LayerNorm fp32.
// Fused attention keeps the full 16x2048 score stripe in LDS (320KB/WGP).
// Round 2: float4-vectorized global staging + packed bf16 conversion; phase-3
// of attention restructured (wave-per-dim-slab, LDS-staged transposed V tile)
// so all WMMA operand traffic is b128 LDS loads.
// ---------------------------------------------------------------------------

typedef __attribute__((ext_vector_type(16))) __bf16 v16bf;
typedef __attribute__((ext_vector_type(8)))  float  v8f;

#define NW   2048   // sequence length (N_WORDS)
#define DIM  512
#define DFF  2048
#define NH   8
#define DK   64

__device__ __forceinline__ __bf16 f2bf(float f) {
  unsigned u = __builtin_bit_cast(unsigned, f);
  u += 0x7fffu + ((u >> 16) & 1u);           // round-to-nearest-even
  unsigned short h = (unsigned short)(u >> 16);
  return __builtin_bit_cast(__bf16, h);
}

// two fp32 -> packed bf16 pair (low = f0, high = f1)
__device__ __forceinline__ unsigned pack2bf(float f0, float f1) {
  unsigned u0 = __builtin_bit_cast(unsigned, f0);
  unsigned u1 = __builtin_bit_cast(unsigned, f1);
  u0 += 0x7fffu + ((u0 >> 16) & 1u);
  u1 += 0x7fffu + ((u1 >> 16) & 1u);
  return (u0 >> 16) | (u1 & 0xffff0000u);
}

__device__ __forceinline__ v8f wmma_bf16(v16bf a, v16bf b, v8f c) {
  return __builtin_amdgcn_wmma_f32_16x16x32_bf16(false, a, false, b, (short)0, c,
                                                 false, false);
}

// ---------------------------------------------------------------------------
// Tiled GEMM: C[M,N] = act(A[M,K] @ W[K,N] + bias [+ res]) ; fp32 in/out,
// bf16 WMMA inner product. Block = 256 threads (8 waves), tile 128x128, K=32.
// M,N multiples of 128; K multiple of 32 (true for all shapes used here).
// ---------------------------------------------------------------------------
#define LDSS 40  // padded LDS row stride in halfs

__global__ void gemm_bf16_k(const float* __restrict__ A, const float* __restrict__ W,
                            const float* __restrict__ bias, const float* __restrict__ res,
                            float* __restrict__ C, int M, int N, int K, int relu) {
  extern __shared__ char smem_raw[];
  __bf16* As = (__bf16*)smem_raw;       // [128][LDSS]  (m-major, k inner)
  __bf16* Bs = As + 128 * LDSS;         // [128][LDSS]  (n-major, k inner)

  const int tid  = threadIdx.x;
  const int lane = tid & 31, wave = tid >> 5;
  const int tileM = blockIdx.x * 128, tileN = blockIdx.y * 128;
  const int m0w = (wave & 3) * 32, n0w = (wave >> 2) * 64;
  const int row = lane & 15, khalf = lane >> 4;

  v8f acc[2][4];
  for (int i = 0; i < 2; ++i)
    for (int j = 0; j < 4; ++j)
      for (int e = 0; e < 8; ++e) acc[i][j][e] = 0.f;

  for (int k0 = 0; k0 < K; k0 += 32) {
    // A tile: 128x32 fp32, one float4 per thread-iteration (coalesced),
    // packed bf16 pair stores (8B aligned).
    for (int i = tid; i < (128 * 32) / 4; i += 256) {
      int r = i >> 3, c = (i & 7) * 4;
      float4 a4 = *(const float4*)&A[(size_t)(tileM + r) * K + k0 + c];
      unsigned* dst = (unsigned*)&As[r * LDSS + c];
      dst[0] = pack2bf(a4.x, a4.y);
      dst[1] = pack2bf(a4.z, a4.w);
    }
    // B tile: 32x128 fp32 loaded as float4 along N (coalesced), stored
    // transposed (n-major) so fragment reads are contiguous b128 runs.
    for (int i = tid; i < (32 * 128) / 4; i += 256) {
      int kk = i >> 5, n = (i & 31) * 4;
      float4 w4 = *(const float4*)&W[(size_t)(k0 + kk) * N + tileN + n];
      Bs[(n + 0) * LDSS + kk] = f2bf(w4.x);
      Bs[(n + 1) * LDSS + kk] = f2bf(w4.y);
      Bs[(n + 2) * LDSS + kk] = f2bf(w4.z);
      Bs[(n + 3) * LDSS + kk] = f2bf(w4.w);
    }
    __syncthreads();

    v16bf afrag[2], bfrag[4];
    for (int mt = 0; mt < 2; ++mt) {
      const __bf16* ap = &As[(m0w + mt * 16 + row) * LDSS + khalf * 8];
      for (int t = 0; t < 8; ++t) afrag[mt][t]     = ap[t];        // K: kb+0..7
      for (int t = 0; t < 8; ++t) afrag[mt][8 + t] = ap[16 + t];   // K: kb+16..23
    }
    for (int nt = 0; nt < 4; ++nt) {
      const __bf16* bp = &Bs[(n0w + nt * 16 + row) * LDSS + khalf * 8];
      for (int t = 0; t < 8; ++t) bfrag[nt][t]     = bp[t];        // K: kb+0..7
      for (int t = 0; t < 8; ++t) bfrag[nt][8 + t] = bp[16 + t];   // K: kb+16..23
    }
    for (int mt = 0; mt < 2; ++mt)
      for (int nt = 0; nt < 4; ++nt)
        acc[mt][nt] = wmma_bf16(afrag[mt], bfrag[nt], acc[mt][nt]);
    __syncthreads();
  }

  for (int mt = 0; mt < 2; ++mt)
    for (int nt = 0; nt < 4; ++nt) {
      int col = tileN + n0w + nt * 16 + row;
      float bv = bias ? bias[col] : 0.f;
      for (int r = 0; r < 8; ++r) {
        int rowg = tileM + m0w + mt * 16 + khalf * 8 + r;
        float v = acc[mt][nt][r] + bv;
        if (res)  v += res[(size_t)rowg * N + col];
        if (relu) v = fmaxf(v, 0.f);
        C[(size_t)rowg * N + col] = v;
      }
    }
}

// ---------------------------------------------------------------------------
// Fused multi-head self-attention (no 1/sqrt(dk) scaling, per reference).
// One block = (head h, 16-query stripe). 128 threads = 4 waves.
// LDS holds the full 16x2048 fp32 score stripe -> single-pass softmax.
// Phase 1: S = Q Kᵀ via WMMA (per-lane contiguous K reads -> b128 loads).
// Phase 3: O = P V; wave w owns dim slab nt=w; V chunks staged in LDS as
//          transposed bf16 so fragments are contiguous b128 LDS loads.
// ---------------------------------------------------------------------------
__global__ void attn_fused_k(const float* __restrict__ Qg, const float* __restrict__ Kg,
                             const float* __restrict__ Vg, float* __restrict__ Og) {
  const int h  = blockIdx.x >> 7;     // blocks = NH * (NW/16)
  const int qb = blockIdx.x & 127;
  const int q0 = qb * 16;

  extern __shared__ char smem_raw[];
  float*  S   = (float*)smem_raw;        // [16][NW]
  float*  red = S + 16 * NW;             // [16][8]
  __bf16* Vs  = (__bf16*)(red + 128);    // [64][LDSS] transposed V tile (dim-major)

  const int tid = threadIdx.x;           // 128
  const int lane = tid & 31, wave = tid >> 5;
  const int row = lane & 15, khalf = lane >> 4;

  // Q fragments for this stripe (K-dim = 64 -> two x32 steps)
  v16bf qf[2];
  for (int s = 0; s < 2; ++s) {
    const float* qp = &Qg[(size_t)(q0 + row) * DIM + h * DK + s * 32 + khalf * 8];
    for (int t = 0; t < 8; ++t) qf[s][t]     = f2bf(qp[t]);
    for (int t = 0; t < 8; ++t) qf[s][8 + t] = f2bf(qp[16 + t]);
  }

  // ---- Phase 1: scores ----
  for (int kt = wave; kt < NW / 16; kt += 4) {
    v8f acc; for (int e = 0; e < 8; ++e) acc[e] = 0.f;
    for (int s = 0; s < 2; ++s) {
      v16bf bf;
      const float* kp = &Kg[(size_t)(kt * 16 + row) * DIM + h * DK + s * 32 + khalf * 8];
      for (int t = 0; t < 8; ++t) bf[t]     = f2bf(kp[t]);
      for (int t = 0; t < 8; ++t) bf[8 + t] = f2bf(kp[16 + t]);
      acc = wmma_bf16(qf[s], bf, acc);
    }
    for (int r = 0; r < 8; ++r) S[(khalf * 8 + r) * NW + kt * 16 + row] = acc[r];
  }
  __syncthreads();

  // ---- Phase 2: softmax over 2048 keys, 8 threads per query row ----
  {
    int r = tid >> 3, st = tid & 7;
    float mx = -3.4e38f;
    for (int j = st; j < NW; j += 8) mx = fmaxf(mx, S[r * NW + j]);
    red[r * 8 + st] = mx; __syncthreads();
    if (st == 0) { float m = red[r * 8]; for (int e = 1; e < 8; ++e) m = fmaxf(m, red[r * 8 + e]); red[r * 8] = m; }
    __syncthreads();
    float m = red[r * 8]; __syncthreads();
    float sum = 0.f;
    for (int j = st; j < NW; j += 8) { float e = expf(S[r * NW + j] - m); S[r * NW + j] = e; sum += e; }
    red[r * 8 + st] = sum; __syncthreads();
    if (st == 0) { float s2 = 0.f; for (int e = 0; e < 8; ++e) s2 += red[r * 8 + e]; red[r * 8] = s2; }
    __syncthreads();
    float inv = 1.f / red[r * 8];
    for (int j = st; j < NW; j += 8) S[r * NW + j] *= inv;
  }
  __syncthreads();

  // ---- Phase 3: O = P V. All 4 waves walk the same 32-key chunks; wave w
  // produces output dims [w*16, w*16+16). V chunk staged as Vs[dim][key].
  v8f oacc; for (int e = 0; e < 8; ++e) oacc[e] = 0.f;
  for (int c = 0; c < NW / 32; ++c) {
    // cooperative stage: 32 keys x 64 dims, float4 coalesced along dims
    for (int i = tid; i < (32 * 64) / 4; i += 128) {
      int key = i >> 4, d = (i & 15) * 4;
      float4 vv = *(const float4*)&Vg[(size_t)(c * 32 + key) * DIM + h * DK + d];
      Vs[(d + 0) * LDSS + key] = f2bf(vv.x);
      Vs[(d + 1) * LDSS + key] = f2bf(vv.y);
      Vs[(d + 2) * LDSS + key] = f2bf(vv.z);
      Vs[(d + 3) * LDSS + key] = f2bf(vv.w);
    }
    __syncthreads();

    v16bf pf;
    const float* pp = &S[row * NW + c * 32 + khalf * 8];
    for (int t = 0; t < 8; ++t) pf[t]     = f2bf(pp[t]);
    for (int t = 0; t < 8; ++t) pf[8 + t] = f2bf(pp[16 + t]);

    v16bf vf;
    const __bf16* vp = &Vs[(wave * 16 + row) * LDSS + khalf * 8];
    for (int t = 0; t < 8; ++t) vf[t]     = vp[t];
    for (int t = 0; t < 8; ++t) vf[8 + t] = vp[16 + t];

    oacc = wmma_bf16(pf, vf, oacc);
    __syncthreads();
  }
  for (int r = 0; r < 8; ++r)
    Og[(size_t)(q0 + khalf * 8 + r) * DIM + h * DK + wave * 16 + row] = oacc[r];
}

// ---------------------------------------------------------------------------
// LayerNorm (unbiased std, /(s+eps), per reference). One block per row.
// ---------------------------------------------------------------------------
__global__ void layernorm_k(const float* __restrict__ in, const float* __restrict__ a,
                            const float* __restrict__ b, float* __restrict__ out, int N) {
  __shared__ float red[256];
  int row = blockIdx.x, tid = threadIdx.x;
  const float* x = in + (size_t)row * N;
  float s = 0.f;
  for (int j = tid; j < N; j += 256) s += x[j];
  red[tid] = s; __syncthreads();
  for (int o = 128; o > 0; o >>= 1) { if (tid < o) red[tid] += red[tid + o]; __syncthreads(); }
  float mean = red[0] / N; __syncthreads();
  float v = 0.f;
  for (int j = tid; j < N; j += 256) { float d = x[j] - mean; v += d * d; }
  red[tid] = v; __syncthreads();
  for (int o = 128; o > 0; o >>= 1) { if (tid < o) red[tid] += red[tid + o]; __syncthreads(); }
  float sd = sqrtf(red[0] / (float)(N - 1)) + 1e-6f;
  for (int j = tid; j < N; j += 256)
    out[(size_t)row * N + j] = a[j] * (x[j] - mean) / sd + b[j];
}

// word embedding + sinusoidal positional encoding
__global__ void embed_pe_k(const int* __restrict__ words, const float* __restrict__ emb,
                           float* __restrict__ x) {
  int idx = blockIdx.x * 256 + threadIdx.x;     // NW*DIM total
  int i = idx >> 9, j = idx & 511;
  float val = emb[(size_t)words[i] * DIM + j];
  float p2 = (float)(j & ~1);
  float ang = (float)i * expf(-p2 * (9.210340371976184f / 512.0f)); // ln(1e4)/d
  val += (j & 1) ? cosf(ang) : sinf(ang);
  x[idx] = val;
}

// GNN compound encoder (64 atoms x 10 dims) -> compound[10]
__global__ void gnn_k(const int* __restrict__ fp, const float* __restrict__ adj,
                      const float* __restrict__ emb,
                      const float* w0, const float* b0, const float* w1, const float* b1,
                      const float* w2, const float* b2, float* __restrict__ compound) {
  __shared__ float xs[64][10], hs[64][10];
  int i = threadIdx.x;                          // 64 threads
  int f = fp[i];
  for (int j = 0; j < 10; ++j) xs[i][j] = emb[(size_t)f * 10 + j];
  __syncthreads();
  const float* Ws[3] = {w0, w1, w2};
  const float* Bs[3] = {b0, b1, b2};
  for (int l = 0; l < 3; ++l) {
    const float* W = Ws[l]; const float* B = Bs[l];
    float hh[10];
    for (int j = 0; j < 10; ++j) {
      float s = B[j];
      for (int k = 0; k < 10; ++k) s += xs[i][k] * W[k * 10 + j];
      hh[j] = fmaxf(s, 0.f);
    }
    for (int j = 0; j < 10; ++j) hs[i][j] = hh[j];
    __syncthreads();
    float upd[10]; for (int j = 0; j < 10; ++j) upd[j] = 0.f;
    for (int k = 0; k < 64; ++k) {
      float a_ = adj[i * 64 + k];
      for (int j = 0; j < 10; ++j) upd[j] += a_ * hs[k][j];
    }
    for (int j = 0; j < 10; ++j) xs[i][j] += upd[j];
    __syncthreads();
  }
  if (i < 10) { float s = 0.f; for (int k = 0; k < 64; ++k) s += xs[k][i]; compound[i] = s / 64.f; }
}

// qvec = l0( tgt_lin(compound) ) : 10 -> 512 -> 512. Block of 512 threads.
__global__ void tgt_query_k(const float* __restrict__ compound,
                            const float* wt, const float* bt,
                            const float* w0, const float* b0, float* __restrict__ qvec) {
  __shared__ float t[512], c[10];
  int tid = threadIdx.x;
  if (tid < 10) c[tid] = compound[tid];
  __syncthreads();
  { float s = bt[tid]; for (int i = 0; i < 10; ++i) s += c[i] * wt[i * DIM + tid]; t[tid] = s; }
  __syncthreads();
  { float s = b0[tid]; for (int i = 0; i < DIM; ++i) s += t[i] * w0[i * DIM + tid]; qvec[tid] = s; }
}

// scores[h][n] = dot(qvec[h*64..], K[n][h*64..])
__global__ void tgt_scores_k(const float* __restrict__ qvec, const float* __restrict__ Kg,
                             float* __restrict__ sc) {
  int idx = blockIdx.x * 256 + threadIdx.x;     // NH*NW
  int h = idx >> 11, key = idx & (NW - 1);
  const float* kp = &Kg[(size_t)key * DIM + h * DK];
  const float* qp = &qvec[h * DK];
  float s = 0.f;
  for (int t = 0; t < DK; ++t) s += qp[t] * kp[t];
  sc[idx] = s;
}

// per-head softmax over 2048 keys
__global__ void softmax_row_k(float* __restrict__ sc, int n) {
  __shared__ float red[256];
  int h = blockIdx.x, tid = threadIdx.x;
  float* r_ = sc + (size_t)h * n;
  float m = -3.4e38f;
  for (int j = tid; j < n; j += 256) m = fmaxf(m, r_[j]);
  red[tid] = m; __syncthreads();
  for (int o = 128; o > 0; o >>= 1) { if (tid < o) red[tid] = fmaxf(red[tid], red[tid + o]); __syncthreads(); }
  m = red[0]; __syncthreads();
  float s = 0.f;
  for (int j = tid; j < n; j += 256) { float e = expf(r_[j] - m); r_[j] = e; s += e; }
  red[tid] = s; __syncthreads();
  for (int o = 128; o > 0; o >>= 1) { if (tid < o) red[tid] += red[tid + o]; __syncthreads(); }
  float inv = 1.f / red[0];
  for (int j = tid; j < n; j += 256) r_[j] *= inv;
}

// decoder tgt-attn: out[n][c] = a[h(c)][n] * V[n][c]   (elementwise scale)
__global__ void tgt_scale_k(const float* __restrict__ sc, const float* __restrict__ Vg,
                            float* __restrict__ outb) {
  int idx = blockIdx.x * 256 + threadIdx.x;     // NW*DIM
  int n2 = idx >> 9, c = idx & 511;
  outb[idx] = sc[(c >> 6) * NW + n2] * Vg[idx];
}

// tgt_out: o[c] = sum_n a[h(c)][n] * V[n][c]
__global__ void tgt_out_reduce_k(const float* __restrict__ sc, const float* __restrict__ Vg,
                                 float* __restrict__ o) {
  int c = threadIdx.x;                          // 512
  int h = c >> 6;
  float s = 0.f;
  for (int n2 = 0; n2 < NW; ++n2) s += sc[h * NW + n2] * Vg[(size_t)n2 * DIM + c];
  o[c] = s;
}

// 512x512 GEMV: out = in @ W + b
__global__ void gemv512_k(const float* __restrict__ in, const float* W, const float* b,
                          float* __restrict__ out) {
  __shared__ float v_[512];
  int tid = threadIdx.x;                        // 512
  v_[tid] = in[tid]; __syncthreads();
  float s = b[tid];
  for (int i = 0; i < DIM; ++i) s += v_[i] * W[i * DIM + tid];
  out[tid] = s;
}

// interaction head: cat(compound, protein) -> 2x (522x522 + relu) -> 522->1
__global__ void head_k(const float* __restrict__ compound, const float* __restrict__ protein,
                       const float* w0, const float* b0, const float* w1, const float* b1,
                       const float* wi, const float* bi, float* __restrict__ out) {
  __shared__ float a_[522], c_[522], red[256];
  int tid = threadIdx.x;                        // 256
  for (int j = tid; j < 522; j += 256) a_[j] = (j < 10) ? compound[j] : protein[j - 10];
  __syncthreads();
  for (int j = tid; j < 522; j += 256) {
    float s = b0[j];
    for (int i = 0; i < 522; ++i) s += a_[i] * w0[i * 522 + j];
    c_[j] = fmaxf(s, 0.f);
  }
  __syncthreads();
  for (int j = tid; j < 522; j += 256) {
    float s = b1[j];
    for (int i = 0; i < 522; ++i) s += c_[i] * w1[i * 522 + j];
    a_[j] = fmaxf(s, 0.f);
  }
  __syncthreads();
  float s = 0.f;
  for (int i = tid; i < 522; i += 256) s += a_[i] * wi[i];
  red[tid] = s; __syncthreads();
  for (int o = 128; o > 0; o >>= 1) { if (tid < o) red[tid] += red[tid + o]; __syncthreads(); }
  if (tid == 0) out[0] = red[0] + bi[0];
}

// ---------------------------------------------------------------------------
extern "C" void kernel_launch(void* const* d_in, const int* in_sizes, int n_in,
                              void* d_out, int out_size, void* d_ws, size_t ws_size,
                              hipStream_t stream) {
  (void)in_sizes; (void)n_in; (void)out_size; (void)ws_size;
  auto F = [&](int i) { return (const float*)d_in[i]; };
  const int* fp    = (const int*)d_in[0];
  const float* adj = F(1);
  const int* words = (const int*)d_in[2];
  const float* embfp = F(3);
  const float* embword = F(10);

  // workspace layout (floats)
  float* ws = (float*)d_ws;
  float* x    = ws;                         // [NW,DIM]
  float* y    = ws + (1u << 20);
  float* qb   = ws + (2u << 20);
  float* kb   = ws + (3u << 20);
  float* vb   = ws + (4u << 20);
  float* attn = ws + (5u << 20);
  float* ffh  = ws + (6u << 20);            // [NW,DFF] (4M floats)
  float* compound = ws + (10u << 20);
  float* qvec = compound + 16;
  float* scores = qvec + 528;               // [NH,NW]
  float* ovec = scores + NH * NW;
  float* ppre = ovec + 512;
  float* prot = ppre + 512;

  const size_t GEMM_LDS = 2u * 128 * LDSS * sizeof(__bf16);              // 20 KB
  const size_t ATT_LDS  = (16u * NW + 128) * 4 + 64u * LDSS * 2;         // ~134 KB

  auto gemm = [&](const float* A, const float* W, const float* bias, const float* res,
                  float* C, int M, int N, int K, int relu) {
    dim3 grid(M / 128, N / 128);
    gemm_bf16_k<<<grid, 256, GEMM_LDS, stream>>>(A, W, bias, res, C, M, N, K, relu);
  };
  auto ln = [&](const float* in, int ia, int ib, float* out, int rows) {
    layernorm_k<<<rows, 256, 0, stream>>>(in, F(ia), F(ib), out, DIM);
  };

  // --- GNN + embeddings ---
  gnn_k<<<1, 64, 0, stream>>>(fp, adj, embfp, F(4), F(5), F(6), F(7), F(8), F(9), compound);
  embed_pe_k<<<(NW * DIM) / 256, 256, 0, stream>>>(words, embword, x);

  // --- encoder: 3 layers ---
  for (int l = 0; l < 3; ++l) {
    int B = 11 + l * 14;  // lq.w/b, lk.w/b, lv.w/b, w1.w/b, w2.w/b, ln0.a/b, ln1.a/b
    gemm(x, F(B + 0), F(B + 1), nullptr, qb, NW, DIM, DIM, 0);
    gemm(x, F(B + 2), F(B + 3), nullptr, kb, NW, DIM, DIM, 0);
    gemm(x, F(B + 4), F(B + 5), nullptr, vb, NW, DIM, DIM, 0);
    attn_fused_k<<<NH * (NW / 16), 128, ATT_LDS, stream>>>(qb, kb, vb, attn);
    gemm(attn, F(B + 4), F(B + 5), x, y, NW, DIM, DIM, 0);   // out-proj reuses lv
    ln(y, B + 10, B + 11, x, NW);
    gemm(x, F(B + 6), F(B + 7), nullptr, ffh, NW, DFF, DIM, 1);
    gemm(ffh, F(B + 8), F(B + 9), x, y, NW, DIM, DFF, 0);
    ln(y, B + 12, B + 13, x, NW);
  }

  // --- decoder: 3 layers ---
  for (int l = 0; l < 3; ++l) {
    int B = 53 + l * 26;
    // ta: tgt.w/b(+0/1) l0(+2/3) l1(+4/5) l2(+6/7) l3(+8/9); self lq/lk/lv(+10..15);
    // ff w1/w2(+16..19); ln0(+20/21) ln1(+22/23) ln2(+24/25)
    tgt_query_k<<<1, 512, 0, stream>>>(compound, F(B + 0), F(B + 1), F(B + 2), F(B + 3), qvec);
    gemm(x, F(B + 4), F(B + 5), nullptr, kb, NW, DIM, DIM, 0);     // l1 -> K
    gemm(x, F(B + 6), F(B + 7), nullptr, vb, NW, DIM, DIM, 0);     // l2 -> V
    tgt_scores_k<<<(NH * NW) / 256, 256, 0, stream>>>(qvec, kb, scores);
    softmax_row_k<<<NH, 256, 0, stream>>>(scores, NW);
    tgt_scale_k<<<(NW * DIM) / 256, 256, 0, stream>>>(scores, vb, attn);
    gemm(attn, F(B + 8), F(B + 9), x, y, NW, DIM, DIM, 0);         // l3 + residual
    ln(y, B + 20, B + 21, x, NW);
    // self-attn
    gemm(x, F(B + 10), F(B + 11), nullptr, qb, NW, DIM, DIM, 0);
    gemm(x, F(B + 12), F(B + 13), nullptr, kb, NW, DIM, DIM, 0);
    gemm(x, F(B + 14), F(B + 15), nullptr, vb, NW, DIM, DIM, 0);
    attn_fused_k<<<NH * (NW / 16), 128, ATT_LDS, stream>>>(qb, kb, vb, attn);
    gemm(attn, F(B + 14), F(B + 15), x, y, NW, DIM, DIM, 0);
    ln(y, B + 22, B + 23, x, NW);
    // ff
    gemm(x, F(B + 16), F(B + 17), nullptr, ffh, NW, DFF, DIM, 1);
    gemm(ffh, F(B + 18), F(B + 19), x, y, NW, DIM, DFF, 0);
    ln(y, B + 24, B + 25, x, NW);
  }

  // --- tgt_out + final LN ---
  tgt_query_k<<<1, 512, 0, stream>>>(compound, F(131), F(132), F(133), F(134), qvec);
  gemm(x, F(135), F(136), nullptr, kb, NW, DIM, DIM, 0);           // l1
  gemm(x, F(137), F(138), nullptr, vb, NW, DIM, DIM, 0);           // l2
  tgt_scores_k<<<(NH * NW) / 256, 256, 0, stream>>>(qvec, kb, scores);
  softmax_row_k<<<NH, 256, 0, stream>>>(scores, NW);
  tgt_out_reduce_k<<<1, 512, 0, stream>>>(scores, vb, ovec);
  gemv512_k<<<1, 512, 0, stream>>>(ovec, F(139), F(140), ppre);    // l3
  layernorm_k<<<1, 256, 0, stream>>>(ppre, F(141), F(142), prot, DIM);

  // --- interaction head ---
  head_k<<<1, 256, 0, stream>>>(compound, prot, F(143), F(144), F(145), F(146),
                                F(147), F(148), (float*)d_out);
}
